// CustomLSTMCell_35476429865299
// MI455X (gfx1250) — matrix operand order repaired and checked by reading, not compiled, not run on hardware
//
#include <hip/hip_runtime.h>
#include <hip/hip_bf16.h>
#include <math.h>

// ---- CDNA5 WMMA types ----
typedef __attribute__((ext_vector_type(16))) __bf16          v16bf;
typedef __attribute__((ext_vector_type(16))) unsigned short  v16us;
typedef __attribute__((ext_vector_type(8)))  unsigned short  v8us;
typedef __attribute__((ext_vector_type(8)))  float           v8f;
typedef __attribute__((ext_vector_type(2)))  float           v2f;
typedef __attribute__((ext_vector_type(2)))  __bf16          v2bf;

#define Hc  1024
#define Bc  8192
#define BM  64
#define BN  128
#define BKt 32
#define LDP 40   // padded LDS row pitch in ushorts: 80B = 20 banks, 16B-aligned rows

// Packed fp32 -> bf16 hi/lo split using hardware conversions (RTE).
__device__ __forceinline__ void split_store(unsigned short* hi, unsigned short* lo, v2f v) {
    v2bf h  = __builtin_convertvector(v, v2bf);
    v2f  hf = __builtin_convertvector(h, v2f);
    v2bf l  = __builtin_convertvector(v - hf, v2bf);
    *(unsigned int*)hi = __builtin_bit_cast(unsigned int, h);
    *(unsigned int*)lo = __builtin_bit_cast(unsigned int, l);
}
// Two aligned 16B LDS loads -> one 16-element bf16 fragment (no scalar packing).
__device__ __forceinline__ v16us ld_frag(const unsigned short* base, int off2) {
    v8us a = *(const v8us*)base;
    v8us b = *(const v8us*)(base + off2);
    return __builtin_shufflevector(a, b, 0,1,2,3,4,5,6,7,8,9,10,11,12,13,14,15);
}
// Fast gate math on hardware TRANS ops: v_exp_f32 + v_rcp_f32.
#define LOG2E 1.4426950408889634f
__device__ __forceinline__ float fast_sigmoid(float xv) {
    return __builtin_amdgcn_rcpf(1.0f + __builtin_amdgcn_exp2f(-LOG2E * xv));
}
__device__ __forceinline__ float fast_tanh(float xv) {
    return 1.0f - 2.0f * __builtin_amdgcn_rcpf(1.0f + __builtin_amdgcn_exp2f(2.0f * LOG2E * xv));
}

// Fused masked-LSTM cell: 8 GEMMs (bf16 hi/lo split, fp32 accumulate) + gate epilogue.
__global__ __launch_bounds__(256, 1)
void lstm_cell_wmma(const float* __restrict__ x,  const float* __restrict__ hx,
                    const float* __restrict__ cx,
                    const float* __restrict__ mask_x, const float* __restrict__ mask_h,
                    const float* __restrict__ Wx_f, const float* __restrict__ Wx_i,
                    const float* __restrict__ Wx_o, const float* __restrict__ Wx_c,
                    const float* __restrict__ Wh_f, const float* __restrict__ Wh_i,
                    const float* __restrict__ Wh_o, const float* __restrict__ Wh_c,
                    const float* __restrict__ bx_f, const float* __restrict__ bx_i,
                    const float* __restrict__ bx_o, const float* __restrict__ bx_c,
                    float* __restrict__ out)
{
    extern __shared__ unsigned short smem[];
    unsigned short* sAhi = smem;                         // [4][BM][LDP]  (A: m-major, k contig)
    unsigned short* sAlo = sAhi + 4 * BM * LDP;
    unsigned short* sBhi = sAlo + 4 * BM * LDP;          // [4][BN][LDP]  (B: n-major, k contig)
    unsigned short* sBlo = sBhi + 4 * BN * LDP;

    const int tid  = threadIdx.x;
    const int lane = tid & 31;
    const int wave = tid >> 5;
    const int wm   = wave & 1;      // 2 waves in M -> 32 rows each
    const int wn   = wave >> 1;     // 4 waves in N -> 32 cols each
    const int bm0  = blockIdx.y * BM;
    const int bn0  = blockIdx.x * BN;

    // gate order g: 0=f, 1=i, 2=o, 3=c  (matches torch.split order of the masks)
    v8f acc[4][2][2];
    #pragma unroll
    for (int g = 0; g < 4; ++g)
        #pragma unroll
        for (int mi = 0; mi < 2; ++mi)
            #pragma unroll
            for (int ni = 0; ni < 2; ++ni)
                acc[g][mi][ni] = (v8f){};

    const float* Wset[2][4] = { { Wx_f, Wx_i, Wx_o, Wx_c },
                                { Wh_f, Wh_i, Wh_o, Wh_c } };

    for (int p = 0; p < 2; ++p) {                 // p=0: x-side, p=1: h-side
        const float* mbase = p ? mask_h : mask_x;
        for (int kk = 0; kk < Hc; kk += BKt) {
            // ---- stage A tiles: masked activations, split hi/lo, packed b32 stores ----
            #pragma unroll
            for (int it = 0; it < (BM * BKt / 2) / 256; ++it) {     // 4 iters
                int idx = it * 256 + tid;
                int m  = idx >> 4;
                int k2 = (idx & 15) * 2;
                int grow = bm0 + m, gk = kk + k2;
                const v2f xv = *(const v2f*)&x[grow * Hc + gk];
                v2f hv = xv;
                if (p) hv = *(const v2f*)&hx[grow * Hc + gk];
                #pragma unroll
                for (int g = 0; g < 4; ++g) {
                    // reference quirk: gate c's h-side linear is applied to x, not hx
                    const v2f sv = (p && g != 3) ? hv : xv;
                    const v2f mv = *(const v2f*)&mbase[grow * 4 * Hc + g * Hc + gk];
                    int o = (g * BM + m) * LDP + k2;
                    split_store(&sAhi[o], &sAlo[o], sv * mv);
                }
            }
            // ---- stage B tiles: B[n][k] = W[n][k] (W stored [out,in] row-major) ----
            #pragma unroll
            for (int g = 0; g < 4; ++g) {
                const float* W = Wset[p][g];
                if (kk + BKt < Hc)
                    __builtin_prefetch(&W[(bn0 + (tid >> 5)) * Hc + kk + BKt], 0, 1);
                #pragma unroll
                for (int it = 0; it < (BKt * BN / 2) / 256; ++it) { // 8 iters
                    int idx = it * 256 + tid;
                    int n  = idx >> 4;
                    int k2 = (idx & 15) * 2;
                    const v2f w = *(const v2f*)&W[(bn0 + n) * Hc + kk + k2];
                    int o = (g * BN + n) * LDP + k2;
                    split_store(&sBhi[o], &sBlo[o], w);
                }
            }
            __syncthreads();

            // ---- WMMA: 4 gates x (2x2 tiles) x 3 split terms ----
            const int MN = lane & 15;
            const int aK = (lane & 16) ? 8  : 0;    // A 16x32 bf16 lane layout
            const int bK = (lane & 16) ? 8  : 0;    // B 32x16 bf16 lane layout
            #pragma unroll
            for (int g = 0; g < 4; ++g) {
                v16us ahiU[2], aloU[2], bhiU[2], bloU[2];
                #pragma unroll
                for (int mi = 0; mi < 2; ++mi) {
                    int o = (g * BM + wm * 32 + mi * 16 + MN) * LDP + aK;
                    ahiU[mi] = ld_frag(&sAhi[o], 16);   // runs at k=aK.. and k=16+aK..
                    aloU[mi] = ld_frag(&sAlo[o], 16);
                }
                #pragma unroll
                for (int ni = 0; ni < 2; ++ni) {
                    int o = (g * BN + wn * 32 + ni * 16 + MN) * LDP + bK;
                    bhiU[ni] = ld_frag(&sBhi[o], 16);   // runs at k=bK.. and k=16+bK..
                    bloU[ni] = ld_frag(&sBlo[o], 16);
                }
                #pragma unroll
                for (int mi = 0; mi < 2; ++mi)
                    #pragma unroll
                    for (int ni = 0; ni < 2; ++ni) {
                        v16bf ah = __builtin_bit_cast(v16bf, ahiU[mi]);
                        v16bf al = __builtin_bit_cast(v16bf, aloU[mi]);
                        v16bf bh = __builtin_bit_cast(v16bf, bhiU[ni]);
                        v16bf bl = __builtin_bit_cast(v16bf, bloU[ni]);
                        acc[g][mi][ni] = __builtin_amdgcn_wmma_f32_16x16x32_bf16(
                            false, ah, false, bh, (short)0, acc[g][mi][ni], false, false);
                        acc[g][mi][ni] = __builtin_amdgcn_wmma_f32_16x16x32_bf16(
                            false, ah, false, bl, (short)0, acc[g][mi][ni], false, false);
                        acc[g][mi][ni] = __builtin_amdgcn_wmma_f32_16x16x32_bf16(
                            false, al, false, bh, (short)0, acc[g][mi][ni], false, false);
                    }
            }
            __syncthreads();
        }
    }

    // ---- epilogue: bias + gates + cell update, direct stores ----
    float* outH = out;
    float* outC = out + (size_t)Bc * Hc;
    const int rBase = bm0 + wm * 32 + ((lane & 16) ? 8 : 0);
    const int cBase = bn0 + wn * 32 + (lane & 15);
    #pragma unroll
    for (int mi = 0; mi < 2; ++mi)
        #pragma unroll
        for (int ni = 0; ni < 2; ++ni) {
            int c = cBase + ni * 16;
            float bf = bx_f[c], bi = bx_i[c], bo = bx_o[c], bc = bx_c[c];
            #pragma unroll
            for (int v = 0; v < 8; ++v) {
                int r = rBase + mi * 16 + v;
                float fg = fast_sigmoid(acc[0][mi][ni][v] + bf);
                float ig = fast_sigmoid(acc[1][mi][ni][v] + bi);
                float og = fast_sigmoid(acc[2][mi][ni][v] + bo);
                float ct = fast_tanh(acc[3][mi][ni][v] + bc);
                float cv = cx[r * Hc + c];
                float cn = fg * cv + ig * ct;
                outC[r * Hc + c] = cn;
                outH[r * Hc + c] = og * fast_tanh(cn);
            }
        }
}

extern "C" void kernel_launch(void* const* d_in, const int* in_sizes, int n_in,
                              void* d_out, int out_size, void* d_ws, size_t ws_size,
                              hipStream_t stream) {
    const float* x      = (const float*)d_in[0];
    const float* hx     = (const float*)d_in[1];
    const float* cx     = (const float*)d_in[2];
    const float* mask_x = (const float*)d_in[3];
    const float* mask_h = (const float*)d_in[4];
    const float* Wx_i   = (const float*)d_in[5];
    const float* Wx_f   = (const float*)d_in[6];
    const float* Wx_c   = (const float*)d_in[7];
    const float* Wx_o   = (const float*)d_in[8];
    const float* Wh_i   = (const float*)d_in[9];
    const float* Wh_f   = (const float*)d_in[10];
    const float* Wh_c   = (const float*)d_in[11];
    const float* Wh_o   = (const float*)d_in[12];
    const float* bx_i   = (const float*)d_in[13];
    const float* bx_f   = (const float*)d_in[14];
    const float* bx_c   = (const float*)d_in[15];
    const float* bx_o   = (const float*)d_in[16];

    dim3 grid(Hc / BN, Bc / BM);   // (8, 128)
    size_t lds = (size_t)(4 * BM * LDP * 2 + 4 * BN * LDP * 2) * sizeof(unsigned short); // 120 KB
    lstm_cell_wmma<<<grid, 256, lds, stream>>>(
        x, hx, cx, mask_x, mask_h,
        Wx_f, Wx_i, Wx_o, Wx_c,
        Wh_f, Wh_i, Wh_o, Wh_c,
        bx_f, bx_i, bx_o, bx_c,
        (float*)d_out);
}
